// MagNet_16269336117436
// MI455X (gfx1250) — matrix operand order repaired and checked by reading, not compiled, not run on hardware
//
#include <hip/hip_runtime.h>
#include <hip/hip_bf16.h>

// Problem constants (from reference)
#define S_LEN   4096
#define N_OBJ   32
#define IN_F    4
#define H_F     64
#define IF_F    8
#define OUT_F   2
#define ROWS    (S_LEN * N_OBJ)          // 131072 encoder rows

typedef __attribute__((ext_vector_type(16))) _Float16 v16h;
typedef __attribute__((ext_vector_type(8)))  float    v8f;

// ---------------------------------------------------------------------------
// WMMA fragment index helpers (per CDNA5 ISA 7.12.2, wave32):
//   A (f16 16x32): lane holds row M = lane&15; element e -> K = e + 8*hi + (e>=8 ? 8 : 0)
//   B (f16 32x16): lane holds col N = lane&15; element e -> K = e + 8*hi + (e>=8 ? 8 : 0)
//   C/D (f32 16x16): lane L, vgpr v -> (M = v + 8*hi, N = lane&15)
// ---------------------------------------------------------------------------
__device__ __forceinline__ int a_k(int e, int hi) { return e + 8 * hi + ((e & 8) ? 8 : 0); }
__device__ __forceinline__ int b_k(int e, int hi) { return e + 8 * hi + ((e & 8) ? 8 : 0); }

__device__ __forceinline__ v8f wmma_f16(v16h a, v16h b, v8f c) {
    return __builtin_amdgcn_wmma_f32_16x16x32_f16(false, a, false, b, (short)0, c,
                                                  false, false);
}

// ---------------------------------------------------------------------------
// gfx1250 hardware tanh (V_TANH_F32 / V_TANH_F16, TRANS class — co-executes
// with WMMA).  Try both plausible builtin spellings; otherwise force the
// instruction via inline asm.  The trailing v_nop satisfies the ISA rule of
// "1 independent op or V_NOP after a TRANS op before its output is used"
// since the compiler cannot schedule around the asm block.
// ---------------------------------------------------------------------------
__device__ __forceinline__ float hw_tanhf(float x) {
#if __has_builtin(__builtin_amdgcn_tanhf)
    return __builtin_amdgcn_tanhf(x);
#elif __has_builtin(__builtin_amdgcn_tanh_f32)
    return __builtin_amdgcn_tanh_f32(x);
#else
    float r;
    asm("v_tanh_f32 %0, %1\n\tv_nop" : "=v"(r) : "v"(x));
    return r;
#endif
}

__device__ __forceinline__ _Float16 hw_tanhh(_Float16 x) {
#if __has_builtin(__builtin_amdgcn_tanh_f16)
    return __builtin_amdgcn_tanh_f16(x);
#elif __has_builtin(__builtin_amdgcn_tanhh)
    return __builtin_amdgcn_tanhh(x);
#else
    return (_Float16)hw_tanhf((float)x);   // cvt + v_tanh_f32 + cvt
#endif
}

// ===========================================================================
// Kernel 1: per-object encoder.
//   h1 = relu(x @ L1W^T + L1b)          (K=4  -> VALU, into LDS as f16)
//   h2 = relu(h1 @ L2W^T + L2b)         (K=64 -> 2x WMMA per 16-col tile)
//   g  = h2 @ I1W^T                     (K=64 -> 2x WMMA per 16-col tile)
//   self[s,i,d] = relu(x @ S1W^T + S1b) . S2W[i,d,:] + S2b[i,d]
// Each wave owns 16 rows; 8 waves / block -> 128 rows / block, 1024 blocks.
// ===========================================================================
__global__ __launch_bounds__(256) void magnet_encode(
    const float* __restrict__ x,
    const float* __restrict__ L1W, const float* __restrict__ L1b,
    const float* __restrict__ L2W, const float* __restrict__ L2b,
    const float* __restrict__ I1W,
    const float* __restrict__ S1W, const float* __restrict__ S1b,
    const float* __restrict__ S2W, const float* __restrict__ S2b,
    _Float16* __restrict__ gws, float* __restrict__ selfws)
{
    __shared__ _Float16 sW2[H_F * H_F];      // L2W as f16, [n][k]
    __shared__ _Float16 sW3[H_F * H_F];      // I1W as f16, [n][k]
    __shared__ _Float16 sH[8][16 * H_F];     // per-wave 16x64 activation tile

    const int tid  = threadIdx.x;
    const int w    = tid >> 5;
    const int lane = tid & 31;
    const int hi   = lane >> 4;
    const int lc   = lane & 15;

    // Stage weight matrices (f32 -> f16) into LDS, shared by all 8 waves.
    for (int idx = tid; idx < H_F * H_F; idx += 256) {
        sW2[idx] = (_Float16)L2W[idx];
        sW3[idx] = (_Float16)I1W[idx];
    }
    __syncthreads();

    const int R0 = blockIdx.x * 128 + w * 16;   // first global row of this wave

    // ---- layer 1 (K=4) on VALU: 2 lanes per row, 32 cols each ----
    {
        const int r  = lane >> 1;
        const int cb = (lane & 1) * 32;
        const float4 xr = *reinterpret_cast<const float4*>(x + (size_t)(R0 + r) * IN_F);
        #pragma unroll
        for (int c = 0; c < 32; ++c) {
            const int col = cb + c;
            float acc = L1b[col]
                      + xr.x * L1W[col * 4 + 0] + xr.y * L1W[col * 4 + 1]
                      + xr.z * L1W[col * 4 + 2] + xr.w * L1W[col * 4 + 3];
            sH[w][r * H_F + col] = (_Float16)fmaxf(acc, 0.0f);
        }
    }
    __syncthreads();

    // ---- layer 2: h2 = relu(h1 @ L2W^T + b2), f16 WMMA, K=64 ----
    v16h aLo, aHi;
    #pragma unroll
    for (int e = 0; e < 16; ++e) {
        const int K = a_k(e, hi);
        aLo[e] = sH[w][lc * H_F + K];
        aHi[e] = sH[w][lc * H_F + 32 + K];
    }
    v8f hacc[4];
    #pragma unroll
    for (int nt = 0; nt < 4; ++nt) {
        const int n = nt * 16 + lc;
        v16h bLo, bHi;
        #pragma unroll
        for (int e = 0; e < 16; ++e) {
            const int K = b_k(e, hi);
            bLo[e] = sW2[n * H_F + K];
            bHi[e] = sW2[n * H_F + 32 + K];
        }
        v8f c = {};
        c = wmma_f16(aLo, bLo, c);
        c = wmma_f16(aHi, bHi, c);
        const float bias = L2b[n];
        #pragma unroll
        for (int v = 0; v < 8; ++v) c[v] = fmaxf(c[v] + bias, 0.0f);
        hacc[nt] = c;
    }
    __syncthreads();

    // write h2 back to LDS (C layout -> row major f16)
    #pragma unroll
    for (int nt = 0; nt < 4; ++nt)
        #pragma unroll
        for (int v = 0; v < 8; ++v)
            sH[w][(v + 8 * hi) * H_F + nt * 16 + lc] = (_Float16)hacc[nt][v];
    __syncthreads();

    // ---- layer 3: g = h2 @ I1W^T (tanh is applied to the *difference* later) ----
    #pragma unroll
    for (int e = 0; e < 16; ++e) {
        const int K = a_k(e, hi);
        aLo[e] = sH[w][lc * H_F + K];
        aHi[e] = sH[w][lc * H_F + 32 + K];
    }
    #pragma unroll
    for (int nt = 0; nt < 4; ++nt) {
        const int n = nt * 16 + lc;
        v16h bLo, bHi;
        #pragma unroll
        for (int e = 0; e < 16; ++e) {
            const int K = b_k(e, hi);
            bLo[e] = sW3[n * H_F + K];
            bHi[e] = sW3[n * H_F + 32 + K];
        }
        v8f c = {};
        c = wmma_f16(aLo, bLo, c);
        c = wmma_f16(aHi, bHi, c);
        #pragma unroll
        for (int v = 0; v < 8; ++v)
            gws[(size_t)(R0 + v + 8 * hi) * H_F + nt * 16 + lc] = (_Float16)c[v];
    }

    // ---- self term (tiny): lanes 0..15 each own one row ----
    if (lane < 16) {
        const int row = R0 + lane;
        const int i   = row & (N_OBJ - 1);
        const float4 xr = *reinterpret_cast<const float4*>(x + (size_t)row * IN_F);
        float hs[4];
        #pragma unroll
        for (int f = 0; f < 4; ++f) {
            float a = S1b[f]
                    + xr.x * S1W[f * 4 + 0] + xr.y * S1W[f * 4 + 1]
                    + xr.z * S1W[f * 4 + 2] + xr.w * S1W[f * 4 + 3];
            hs[f] = fmaxf(a, 0.0f);
        }
        #pragma unroll
        for (int d = 0; d < 2; ++d) {
            selfws[(size_t)row * 2 + d] =
                S2b[i * 2 + d]
                + hs[d * 2 + 0] * S2W[i * 4 + d * 2 + 0]
                + hs[d * 2 + 1] * S2W[i * 4 + d * 2 + 1];
        }
    }
}

// ===========================================================================
// Kernel 2: pairwise interaction.  One block per sequence step s.
//   pair  = tanh(g_i - g_j)                        (inner tanh, V_TANH)
//   feat  = tanh(pair @ I2W^T)                     (WMMA + outer V_TANH_F32)
//   out[s,i,d] = self[s,i,d] + sum_{j,k} feat[d*4+k] * I3d[i,j,d,k]
// Tile = 16 pairs (fixed i, j in [j0, j0+16)) x K=64 -> 2 WMMAs, B = I2W^T
// zero-padded 8->16 cols.  Wave w owns i = 4w..4w+3 -> no output races.
// ===========================================================================
__global__ __launch_bounds__(256) void magnet_pairs(
    const _Float16* __restrict__ gws,
    const float* __restrict__ selfws,
    const float* __restrict__ I2W,
    const float* __restrict__ I3,
    float* __restrict__ out)
{
    __shared__ _Float16 sG[N_OBJ * H_F];     // 32 x 64 f16 = 4 KB

    const int s    = blockIdx.x;
    const int tid  = threadIdx.x;
    const int w    = tid >> 5;
    const int lane = tid & 31;
    const int hi   = lane >> 4;
    const int lc   = lane & 15;

    // stage g[s] into LDS: 4 KB via one 16B load per thread
    {
        const uint4* src = reinterpret_cast<const uint4*>(gws + (size_t)s * N_OBJ * H_F);
        reinterpret_cast<uint4*>(sG)[tid] = src[tid];
    }
    __syncthreads();

    // B fragments for I2W^T (64x16, cols 8..15 are zero padding), kept in regs
    v16h bLo, bHi;
    #pragma unroll
    for (int e = 0; e < 16; ++e) {
        const int K = b_k(e, hi);
        bLo[e] = (lc < IF_F) ? (_Float16)I2W[lc * H_F + K]      : (_Float16)0.0f;
        bHi[e] = (lc < IF_F) ? (_Float16)I2W[lc * H_F + 32 + K] : (_Float16)0.0f;
    }

    const int col = lc;                       // this lane's C column (feature c)

    #pragma unroll
    for (int il = 0; il < 4; ++il) {
        const int i = w * 4 + il;
        float partial = 0.0f;

        #pragma unroll
        for (int jh = 0; jh < 2; ++jh) {
            const int j0 = jh * 16;
            const int jA = j0 + lc;           // A-matrix row -> pair (i, jA)

            // A fragment = tanh(g_i - g_j) in f16 (inner edge-MLP tanh)
            v16h aLo, aHi;
            #pragma unroll
            for (int e = 0; e < 16; ++e) {
                const int K = a_k(e, hi);
                aLo[e] = hw_tanhh(sG[i * H_F + K]      - sG[jA * H_F + K]);
                aHi[e] = hw_tanhh(sG[i * H_F + 32 + K] - sG[jA * H_F + 32 + K]);
            }

            v8f c = {};
            c = wmma_f16(aLo, bLo, c);
            c = wmma_f16(aHi, bHi, c);

            // outer tanh + weighted reduce against scattered I3 (diag weight = 0)
            #pragma unroll
            for (int v = 0; v < 8; ++v) {
                const int j = j0 + v + 8 * hi;        // C row -> pair (i, j)
                float wgt = 0.0f;
                if (col < IF_F && j != i) {
                    const int pidx = i * (N_OBJ - 1) + j - (j > i ? 1 : 0);
                    wgt = I3[pidx * IF_F + col];
                }
                partial += hw_tanhf(c[v]) * wgt;
            }
        }

        // cols 0..3 -> d=0, cols 4..7 -> d=1; cols 8..15 carry 0.
        partial += __shfl_xor(partial, 1,  32);
        partial += __shfl_xor(partial, 2,  32);
        partial += __shfl_xor(partial, 8,  32);
        partial += __shfl_xor(partial, 16, 32);

        const size_t ro = ((size_t)s * N_OBJ + i) * OUT_F;
        if (lane == 0) out[ro + 0] = selfws[ro + 0] + partial;
        if (lane == 4) out[ro + 1] = selfws[ro + 1] + partial;
    }
}

// ===========================================================================
extern "C" void kernel_launch(void* const* d_in, const int* in_sizes, int n_in,
                              void* d_out, int out_size, void* d_ws, size_t ws_size,
                              hipStream_t stream)
{
    const float* x   = (const float*)d_in[0];
    const float* L1W = (const float*)d_in[1];
    const float* L1b = (const float*)d_in[2];
    const float* L2W = (const float*)d_in[3];
    const float* L2b = (const float*)d_in[4];
    const float* I1W = (const float*)d_in[5];
    const float* I2W = (const float*)d_in[6];
    const float* I3  = (const float*)d_in[7];
    const float* S1W = (const float*)d_in[8];
    const float* S1b = (const float*)d_in[9];
    const float* S2W = (const float*)d_in[10];
    const float* S2b = (const float*)d_in[11];
    float* out = (float*)d_out;

    // workspace: g (f16, 16 MB) then self term (f32, 1 MB)
    _Float16* gws    = (_Float16*)d_ws;
    float*    selfws = (float*)((char*)d_ws + (size_t)ROWS * H_F * sizeof(_Float16));

    magnet_encode<<<ROWS / 128, 256, 0, stream>>>(x, L1W, L1b, L2W, L2b, I1W,
                                                  S1W, S1b, S2W, S2b, gws, selfws);
    magnet_pairs<<<S_LEN, 256, 0, stream>>>(gws, selfws, I2W, I3, out);
}
